// PerformerAttention_80307298501129
// MI455X (gfx1250) — matrix operand order repaired and checked by reading, not compiled, not run on hardware
//
#include <hip/hip_runtime.h>
#include <hip/hip_bf16.h>

typedef __bf16 bf16;
typedef __attribute__((ext_vector_type(16))) __bf16 bf16x16;
typedef __attribute__((ext_vector_type(8)))  __bf16 bf16x8;
typedef __attribute__((ext_vector_type(4)))  __bf16 bf16x4;
typedef __attribute__((ext_vector_type(8)))  float  v8f;

static constexpr int BB = 2;
static constexpr int LL = 4096;
static constexpr int DM = 512;
static constexpr int HH = 8;
static constexpr int DH = 64;
static constexpr int MF = 266;   // real feature count
static constexpr int MP = 288;   // padded to multiple of 32 (18 x 16)
static constexpr int NE = 80;    // extended N: 64 outputs + 16 (col 64 = den/z)

// CDNA5 async global->LDS path (ASYNCcnt), guarded: falls back to synchronous
// loads if this toolchain doesn't expose the builtins.  (Host pass always takes
// the fallback branch; only LDS layout differs, host just allocates the max.)
#if defined(__AMDGCN__) &&                                            \
    __has_builtin(__builtin_amdgcn_global_load_async_to_lds_b128) &&  \
    __has_builtin(__builtin_amdgcn_s_wait_asynccnt)
#define USE_ASYNC 1
#else
#define USE_ASYNC 0
#endif

__device__ __forceinline__ v8f zero8() {
  v8f z = {0.f,0.f,0.f,0.f,0.f,0.f,0.f,0.f};
  return z;
}

__device__ __forceinline__ v8f wmma_bf(bf16x16 a, bf16x16 b, v8f c) {
  return __builtin_amdgcn_wmma_f32_16x16x32_bf16(false, a, false, b, (short)0, c,
                                                 false, false);
}

// A fragment (16x32 bf16): lane holds row = lane%16; elements 0..7 = K[klo..klo+7],
// 8..15 = K[klo+16..klo+23], klo = (lane/16)*8.  `rowk` points at row start + kb*32.
__device__ __forceinline__ bf16x16 ldsA(const bf16* rowk, int lane) {
  const int klo = (lane >> 4) << 3;
  bf16x8 lo = *(const bf16x8*)(rowk + klo);
  bf16x8 hi = *(const bf16x8*)(rowk + klo + 16);
  return __builtin_shufflevector(lo, hi, 0,1,2,3,4,5,6,7,8,9,10,11,12,13,14,15);
}

// B fragment (32x16 bf16): lane holds column = lane%16; elements 0..7 = K[klo..klo+7],
// 8..15 = K[klo+16..klo+23], klo = (lane/16)*8.  `colk` = column-major column start + kb*32.
__device__ __forceinline__ bf16x16 ldsB(const bf16* colk, int lane) {
  const int klo = (lane >> 4) << 3;
  bf16x8 lo = *(const bf16x8*)(colk + klo);
  bf16x8 hi = *(const bf16x8*)(colk + klo + 16);
  return __builtin_shufflevector(lo, hi, 0,1,2,3,4,5,6,7,8,9,10,11,12,13,14,15);
}

#if USE_ASYNC
// Builtin signature (from compiler diagnostic): first param is a global
// (addrspace(1), printed '__device__') pointer to int __vector_size__(16).
typedef int v4i __attribute__((vector_size(16)));
typedef __attribute__((address_space(1))) void* as1_void_p;
typedef __attribute__((address_space(3))) void* as3_void_p;
typedef __attribute__((address_space(1))) v4i* as1_v4i_p;
typedef __attribute__((address_space(3))) v4i* as3_v4i_p;

// Per-lane 16B async copy, tracked by ASYNCcnt; no VGPR data staging.
__device__ __forceinline__ void async16(const bf16* gsrc, bf16* ldst) {
  __builtin_amdgcn_global_load_async_to_lds_b128(
      (as1_v4i_p)(as1_void_p)(void*)gsrc,
      (as3_v4i_p)(as3_void_p)(void*)ldst, 0, 0);
}
#endif

// ---------------------------------------------------------------------------
// Kernel 1: X @ W^T (+bias, scale), then optionally random-feature projection
// + exp kernel.  mode 0=q', 1=k' (write [B,H,L,MP] bf16), 2=v (write [B,H,L,DH]).
// Grid: (B*L/64, H), 256 threads (8 waves).
// ---------------------------------------------------------------------------
__global__ void __launch_bounds__(256, 1)
proj_feat_kernel(const float* __restrict__ X, const float* __restrict__ W,
                 const float* __restrict__ bias, const float* __restrict__ RF,
                 bf16* __restrict__ out, int mode, float inv_scale) {
  extern __shared__ __align__(256) char smem[];
  bf16* xlds = (bf16*)smem;               // [64][512]
  bf16* wlds = (bf16*)(smem + 65536);     // [64][512]  W rows h*64.. (row-major = B col-major)
  bf16* g    = (bf16*)(smem + 131072);    // [64][64]   head-projected tile
  bf16* rf   = (bf16*)(smem + 139264);    // [288][64]  RF row-major (= B col-major), padded 0
  float* hq  = (float*)(smem + 176128);   // [64]

  const int t = blockIdx.x;
  const int h = blockIdx.y;
  const int tid = threadIdx.x;
  const int w = tid >> 5;
  const int lane = tid & 31;
  const int colq = lane & 15;
  const int rhalf = (lane >> 4) * 8;

  // --- load X tile + W slab (f32 -> bf16), RF, zero hq -----------------------
  {
    const float4* xs = (const float4*)(X + (size_t)t * 64 * DM);
    const float4* ws = (const float4*)(W + (size_t)h * DH * DM);
    for (int i = tid; i < 64 * DM / 4; i += 256) {
      float4 f = xs[i];
      bf16x4 o = {(bf16)f.x, (bf16)f.y, (bf16)f.z, (bf16)f.w};
      *(bf16x4*)(xlds + i * 4) = o;
      float4 fw = ws[i];
      bf16x4 o2 = {(bf16)fw.x, (bf16)fw.y, (bf16)fw.z, (bf16)fw.w};
      *(bf16x4*)(wlds + i * 4) = o2;
    }
    for (int i = tid; i < MF * DH; i += 256) rf[i] = (bf16)RF[i];
    for (int i = tid; i < (MP - MF) * DH; i += 256) rf[MF * DH + i] = (bf16)0.0f;
    if (tid < 64) hq[tid] = 0.0f;
  }
  __syncthreads();

  // --- stage 1: 64x64 head tile = X[64x512] @ W^T slab -----------------------
  const int rt = w & 3;
  const int ctbase = (w >> 2) * 2;
  v8f acc[2] = {zero8(), zero8()};
  for (int kb = 0; kb < DM / 32; kb++) {
    bf16x16 a = ldsA(xlds + (rt * 16 + colq) * DM + kb * 32, lane);
#pragma unroll
    for (int c2 = 0; c2 < 2; c2++) {
      bf16x16 b = ldsB(wlds + ((ctbase + c2) * 16 + colq) * DM + kb * 32, lane);
      acc[c2] = wmma_bf(a, b, acc[c2]);
    }
  }

#pragma unroll
  for (int c2 = 0; c2 < 2; c2++) {
    const int colg = (ctbase + c2) * 16 + colq;
    const float bval = bias[h * DH + colg];
#pragma unroll
    for (int r = 0; r < 8; r++) {
      const int rowg = rt * 16 + rhalf + r;
      const float val = (acc[c2][r] + bval) * inv_scale;
      if (mode == 2) {
        const int flat = t * 64 + rowg;
        const int bb = flat >> 12, l = flat & (LL - 1);
        out[((size_t)(bb * HH + h) * LL + l) * DH + colg] = (bf16)val;
      } else {
        atomicAdd(&hq[rowg], -0.5f * val * val);
        g[rowg * DH + colg] = (bf16)val;
      }
    }
  }
  if (mode == 2) return;
  __syncthreads();

  // --- stage 2: q' = exp(hq + g @ RF^T), padded features forced to 0 ---------
  v8f p[9];
#pragma unroll
  for (int j = 0; j < 9; j++) p[j] = zero8();
  const int par = (w >> 2);
#pragma unroll
  for (int kb = 0; kb < 2; kb++) {
    bf16x16 a = ldsA(g + (rt * 16 + colq) * DH + kb * 32, lane);
#pragma unroll
    for (int j = 0; j < 9; j++) {
      const int ctf = j * 2 + par;
      bf16x16 b = ldsB(rf + (ctf * 16 + colq) * DH + kb * 32, lane);
      p[j] = wmma_bf(a, b, p[j]);
    }
  }
#pragma unroll
  for (int j = 0; j < 9; j++) {
    const int m = (j * 2 + par) * 16 + colq;
#pragma unroll
    for (int r = 0; r < 8; r++) {
      const int rowg = rt * 16 + rhalf + r;
      const int flat = t * 64 + rowg;
      const int bb = flat >> 12, l = flat & (LL - 1);
      const float val = (m < MF) ? __expf(hq[rowg] + p[j][r]) : 0.0f;
      out[((size_t)(bb * HH + h) * LL + l) * (size_t)MP + m] = (bf16)val;
    }
  }
}

// ---------------------------------------------------------------------------
// Kernel 2: chunked causal FAVOR+ scan.  One block per (b,h); 5 waves.
// State S_ext[288x80] f32 lives in VGPR fragments (18/wave), bf16 shadow sT in
// LDS (column-major) is the B matrix of the cross term.  Column 64 of the
// extended state/outputs carries z / denominator (vT col 64 = ones).
// Async path: ping-pong q/k/v chunk buffers filled by GLOBAL_LOAD_ASYNC_TO_LDS
// one chunk ahead, hiding global latency behind the WMMA stages.
// Ping-pong selection is pure offset arithmetic (no LDS-pointer arrays: those
// become addrspacecast static initializers that ld.lld rejects).
// ---------------------------------------------------------------------------
__global__ void __launch_bounds__(160, 1)
scan_kernel(const bf16* __restrict__ qp, const bf16* __restrict__ kp,
            const bf16* __restrict__ vh, bf16* __restrict__ ctx) {
  extern __shared__ __align__(256) char smem[];
#if USE_ASYNC
  // layout: q0 @0, q1 @36864, k0 @73728, k1 @110592, v0 @147456, v1 @155648
  bf16* kT   = (bf16*)(smem + 163840);    // [288][64]
  bf16* vT   = (bf16*)(smem + 200704);    // [80][64]
  bf16* att  = (bf16*)(smem + 210944);    // [64][64]
  bf16* sT   = (bf16*)(smem + 219136);    // [80][288]
  float* den = (float*)(smem + 265216);   // [64]
  bf16 *qlds = nullptr, *klds = nullptr;
#else
  bf16* qlds = (bf16*)smem;               // [64][288] q' chunk (row-major, A)
  bf16* klds = (bf16*)(smem + 36864);     // [64][288] k' chunk (B for Q'K'^T)
  bf16* kT   = (bf16*)(smem + 73728);     // [288][64] k' transposed (A for K'^T V)
  bf16* vT   = (bf16*)(smem + 110592);    // [80][64]  V_ext col-major (B); row64=1s
  bf16* att  = (bf16*)(smem + 120832);    // [64][64]  masked A (A for att@V)
  bf16* sT   = (bf16*)(smem + 129024);    // [80][288] S_ext col-major bf16 (B)
  float* den = (float*)(smem + 175104);   // [64]
#endif

  const int bid = blockIdx.x;
  const int b = bid / HH, h = bid % HH;
  const int tid = threadIdx.x;
  const int w = tid >> 5, lane = tid & 31;
  const int colq = lane & 15, rhalf = (lane >> 4) * 8;

  const bf16* qbase = qp + (size_t)(b * HH + h) * LL * MP;
  const bf16* kbase = kp + (size_t)(b * HH + h) * LL * MP;
  const bf16* vbase = vh + (size_t)(b * HH + h) * LL * DH;

#if USE_ASYNC
  // issue chunk 0 as early as possible (into ping buffers, cur=0)
  {
    bf16* q0 = (bf16*)smem;
    bf16* k0 = (bf16*)(smem + 73728);
    bf16* v0 = (bf16*)(smem + 147456);
    for (int i = tid; i < 64 * MP / 8; i += 160) {
      async16(qbase + i * 8, q0 + i * 8);
      async16(kbase + i * 8, k0 + i * 8);
    }
    for (int i = tid; i < 64 * DH / 8; i += 160)
      async16(vbase + i * 8, v0 + i * 8);
  }
  int cur = 0;
#endif

  for (int i = tid; i < NE * MP / 2; i += 160) ((unsigned int*)sT)[i] = 0u;
  for (int i = tid; i < NE * DH / 2; i += 160) ((unsigned int*)vT)[i] = 0u;
  __syncthreads();
  if (tid < 64) vT[64 * DH + tid] = (bf16)1.0f;  // ones column -> z / den

  v8f S[18];
#pragma unroll
  for (int i = 0; i < 18; i++) S[i] = zero8();

  for (int ci = 0; ci < LL / 64; ci++) {
    const int l0 = ci * 64;

#if USE_ASYNC
    __builtin_amdgcn_s_wait_asynccnt(0);  // chunk ci landed in buf[cur]
    __syncthreads();                      // all waves' copies + prev compute done
    qlds = (bf16*)(smem + cur * 36864);
    klds = (bf16*)(smem + 73728 + cur * 36864);
    bf16* vbuf = (bf16*)(smem + 147456 + cur * 8192);
    if (ci + 1 < LL / 64) {               // stream chunk ci+1 into other buffer
      const int nb = cur ^ 1;
      const bf16* qn = qbase + (size_t)(l0 + 64) * MP;
      const bf16* kn = kbase + (size_t)(l0 + 64) * MP;
      const bf16* vn = vbase + (size_t)(l0 + 64) * DH;
      bf16* qd = (bf16*)(smem + nb * 36864);
      bf16* kd = (bf16*)(smem + 73728 + nb * 36864);
      bf16* vd = (bf16*)(smem + 147456 + nb * 8192);
      for (int i = tid; i < 64 * MP / 8; i += 160) {
        async16(qn + i * 8, qd + i * 8);
        async16(kn + i * 8, kd + i * 8);
      }
      for (int i = tid; i < 64 * DH / 8; i += 160)
        async16(vn + i * 8, vd + i * 8);
    }
    cur ^= 1;
    // build transposed copies from LDS (overlaps with in-flight async stream)
    for (int i = tid; i < 64 * MP / 8; i += 160) {
      bf16x8 kv = *(const bf16x8*)(klds + i * 8);
      const int c = (i * 8) / MP, m0 = (i * 8) % MP;
#pragma unroll
      for (int j = 0; j < 8; j++) kT[(m0 + j) * DH + c] = kv[j];
    }
    for (int i = tid; i < 64 * DH / 8; i += 160) {
      bf16x8 vv = *(const bf16x8*)(vbuf + i * 8);
      const int c = (i * 8) / DH, d0 = (i * 8) % DH;
#pragma unroll
      for (int j = 0; j < 8; j++) vT[(d0 + j) * DH + c] = vv[j];
    }
#else
    __syncthreads();  // prev chunk compute done; sT from prev chunk visible
    const bf16x8* qs = (const bf16x8*)(qbase + (size_t)l0 * MP);
    const bf16x8* ks = (const bf16x8*)(kbase + (size_t)l0 * MP);
    const bf16x8* vs = (const bf16x8*)(vbase + (size_t)l0 * DH);
    for (int i = tid; i < 64 * MP / 8; i += 160) ((bf16x8*)qlds)[i] = qs[i];
    for (int i = tid; i < 64 * MP / 8; i += 160) {
      bf16x8 kv = ks[i];
      ((bf16x8*)klds)[i] = kv;
      const int c = (i * 8) / MP, m0 = (i * 8) % MP;
#pragma unroll
      for (int j = 0; j < 8; j++) kT[(m0 + j) * DH + c] = kv[j];
    }
    for (int i = tid; i < 64 * DH / 8; i += 160) {
      bf16x8 vv = vs[i];
      const int c = (i * 8) / DH, d0 = (i * 8) % DH;
#pragma unroll
      for (int j = 0; j < 8; j++) vT[(d0 + j) * DH + c] = vv[j];
    }
    if (ci + 1 < LL / 64) {  // hint next chunk into cache (global_prefetch_b8)
      __builtin_prefetch(qbase + (size_t)(l0 + 64) * MP + lane * 64, 0, 0);
      __builtin_prefetch(kbase + (size_t)(l0 + 64) * MP + lane * 64, 0, 0);
    }
#endif
    __syncthreads();

    // ---- matmul 1&2: num_ext = Q' @ S_ext ; A_att = Q' @ K'^T (waves 0-3) ----
    v8f num[4], attf[4];
#pragma unroll
    for (int r4 = 0; r4 < 4; r4++) { num[r4] = zero8(); attf[r4] = zero8(); }
    for (int kb = 0; kb < MP / 32; kb++) {
      bf16x16 bS = ldsB(sT + (w * 16 + colq) * MP + kb * 32, lane);
      bf16x16 bK = ldsB(klds + ((w & 3) * 16 + colq) * MP + kb * 32, lane);
#pragma unroll
      for (int rt = 0; rt < 4; rt++) {
        bf16x16 a = ldsA(qlds + (rt * 16 + colq) * MP + kb * 32, lane);
        num[rt] = wmma_bf(a, bS, num[rt]);
        if (w < 4) attf[rt] = wmma_bf(a, bK, attf[rt]);
      }
    }

    // ---- causal mask (s <= t) + bf16 store of A ----
    if (w < 4) {
#pragma unroll
      for (int rt = 0; rt < 4; rt++)
#pragma unroll
        for (int r = 0; r < 8; r++) {
          const int rowg = rt * 16 + rhalf + r;
          const int colg = w * 16 + colq;
          const float av = (rowg >= colg) ? attf[rt][r] : 0.0f;
          att[rowg * 64 + colg] = (bf16)av;
        }
    }
    __syncthreads();  // att visible; all waves done reading sT

    // ---- matmul 3&4: num_ext += A_masked @ V_ext ; S_ext += K'^T @ V_ext ----
    for (int kb = 0; kb < 2; kb++) {
      bf16x16 bV = ldsB(vT + (w * 16 + colq) * DH + kb * 32, lane);
#pragma unroll
      for (int rt = 0; rt < 4; rt++) {
        bf16x16 a = ldsA(att + (rt * 16 + colq) * 64 + kb * 32, lane);
        num[rt] = wmma_bf(a, bV, num[rt]);
      }
#pragma unroll
      for (int mt = 0; mt < 18; mt++) {
        bf16x16 a = ldsA(kT + (mt * 16 + colq) * DH + kb * 32, lane);
        S[mt] = wmma_bf(a, bV, S[mt]);
      }
    }

    // ---- den extraction (wave 4, column 64) + refresh bf16 state shadow ----
    if (w == 4 && colq == 0) {
#pragma unroll
      for (int rt = 0; rt < 4; rt++)
#pragma unroll
        for (int r = 0; r < 8; r++) den[rt * 16 + rhalf + r] = num[rt][r];
    }
#pragma unroll
    for (int mt = 0; mt < 18; mt++)
#pragma unroll
      for (int r = 0; r < 8; r++)
        sT[(w * 16 + colq) * MP + mt * 16 + rhalf + r] = (bf16)S[mt][r];
    __syncthreads();  // den visible

    // ---- output: ctx[b][l][h*64+d] = num / den ----
    if (w < 4) {
#pragma unroll
      for (int rt = 0; rt < 4; rt++)
#pragma unroll
        for (int r = 0; r < 8; r++) {
          const int rowg = rt * 16 + rhalf + r;
          const int colg = w * 16 + colq;
          const float o = num[rt][r] / den[rowg];
          ctx[((size_t)b * LL + l0 + rowg) * DM + h * DH + colg] = (bf16)o;
        }
    }
  }
}

// ---------------------------------------------------------------------------
// Kernel 3: out = ctx @ Wout^T + bout  (f32 output).  Grid (B*L/64, 8).
// ---------------------------------------------------------------------------
__global__ void __launch_bounds__(256, 1)
outproj_kernel(const bf16* __restrict__ ctxb, const float* __restrict__ Wout,
               const float* __restrict__ bout, float* __restrict__ out) {
  extern __shared__ __align__(256) char smem[];
  bf16* xlds = (bf16*)smem;             // [64][512]
  bf16* wlds = (bf16*)(smem + 65536);   // [64][512]
  const int t = blockIdx.x, cs = blockIdx.y;
  const int tid = threadIdx.x, w = tid >> 5, lane = tid & 31;
  const int colq = lane & 15, rhalf = (lane >> 4) * 8;

  const bf16x8* xs = (const bf16x8*)(ctxb + (size_t)t * 64 * DM);
  for (int i = tid; i < 64 * DM / 8; i += 256) ((bf16x8*)xlds)[i] = xs[i];
  const float4* wsrc = (const float4*)(Wout + (size_t)cs * 64 * DM);
  for (int i = tid; i < 64 * DM / 4; i += 256) {
    float4 f = wsrc[i];
    bf16x4 o = {(bf16)f.x, (bf16)f.y, (bf16)f.z, (bf16)f.w};
    *(bf16x4*)(wlds + i * 4) = o;
  }
  __syncthreads();

  const int rt = w & 3, ctbase = (w >> 2) * 2;
  v8f acc[2] = {zero8(), zero8()};
  for (int kb = 0; kb < DM / 32; kb++) {
    bf16x16 a = ldsA(xlds + (rt * 16 + colq) * DM + kb * 32, lane);
#pragma unroll
    for (int c2 = 0; c2 < 2; c2++) {
      bf16x16 b = ldsB(wlds + ((ctbase + c2) * 16 + colq) * DM + kb * 32, lane);
      acc[c2] = wmma_bf(a, b, acc[c2]);
    }
  }
#pragma unroll
  for (int c2 = 0; c2 < 2; c2++) {
    const int colg = (ctbase + c2) * 16 + colq;
    const float bv = bout[cs * 64 + colg];
#pragma unroll
    for (int r = 0; r < 8; r++) {
      const int rowg = rt * 16 + rhalf + r;
      out[((size_t)t * 64 + rowg) * DM + cs * 64 + colg] = acc[c2][r] + bv;
    }
  }
}

// ---------------------------------------------------------------------------
extern "C" void kernel_launch(void* const* d_in, const int* in_sizes, int n_in,
                              void* d_out, int out_size, void* d_ws, size_t ws_size,
                              hipStream_t stream) {
  (void)in_sizes; (void)n_in; (void)out_size; (void)ws_size;
  const float* query = (const float*)d_in[0];
  const float* key   = (const float*)d_in[1];
  const float* value = (const float*)d_in[2];
  const float* Wq    = (const float*)d_in[3];
  const float* bq    = (const float*)d_in[4];
  const float* Wk    = (const float*)d_in[5];
  const float* bk    = (const float*)d_in[6];
  const float* Wv    = (const float*)d_in[7];
  const float* bv    = (const float*)d_in[8];
  const float* Wout  = (const float*)d_in[9];
  const float* bout  = (const float*)d_in[10];
  const float* RF    = (const float*)d_in[11];

  char* ws = (char*)d_ws;
  // qp/kp: [B][H][L][MP] bf16 = 37,748,736 B each; vh: [B][H][L][DH] bf16;
  // ctx: [B][L][DM] bf16.
  bf16* qp  = (bf16*)ws;
  bf16* kp  = (bf16*)(ws + (size_t)37748736);
  bf16* vhb = (bf16*)(ws + (size_t)75497472);
  bf16* ctx = (bf16*)(ws + (size_t)83886080);

  const float inv_scale = 0.21022410381342864f;  // 512^(-1/4)

  dim3 gp(BB * LL / 64, HH);
  proj_feat_kernel<<<gp, 256, 176384, stream>>>(query, Wq, bq, RF, qp, 0, inv_scale);
  proj_feat_kernel<<<gp, 256, 176384, stream>>>(key,   Wk, bk, RF, kp, 1, inv_scale);
  proj_feat_kernel<<<gp, 256, 176384, stream>>>(value, Wv, bv, RF, vhb, 2, 1.0f);
  // scan: allocate the async double-buffer layout size (covers fallback too)
  scan_kernel<<<BB * HH, 160, 265472, stream>>>(qp, kp, vhb, ctx);
  outproj_kernel<<<dim3(BB * LL / 64, HH), 256, 131072, stream>>>(ctx, Wout, bout,
                                                                  (float*)d_out);
}
